// CrossModalAttention_1614907703840
// MI455X (gfx1250) — compile-verified, matrix operand order checked by reading, not run on hardware
//
#include <hip/hip_runtime.h>

// ---------------------------------------------------------------------------
// CrossModalAttention on MI455X (gfx1250, wave32, WMMA + TDM).
//
// Shapes: B=4, S=4096, T=2048, SD=512, TD=768, H=8, D=64.
// ~99 GFLOP, ~110 MB traffic -> compute-bound => all matmuls through
// v_wmma_f32_16x16x32_bf16 (bf16 A/B, f32 accum). Attention K/V chunks are
// DMA'd into LDS by the Tensor Data Mover (double-buffered, TENSORcnt),
// shared by all 8 waves of the block.
// ---------------------------------------------------------------------------

#define B_  4
#define S_  4096
#define T_  2048
#define SD_ 512
#define TD_ 768
#define H_  8
#define D_  64

typedef __bf16 bf16;
typedef unsigned int u32;
typedef __attribute__((ext_vector_type(16))) __bf16 v16bf;
typedef __attribute__((ext_vector_type(8)))  __bf16 v8bf;
typedef __attribute__((ext_vector_type(8)))  float  v8f;
typedef __attribute__((ext_vector_type(4)))  u32    v4u;
typedef __attribute__((ext_vector_type(8)))  int    v8i;
typedef __attribute__((ext_vector_type(4)))  int    v4i;

union Frag16 {
  v16bf v;
  struct { v8bf lo, hi; } h;
};

// A/B fragment loader for 16x32 bf16 WMMA operand, per ISA VGPR layout:
// lane (low = lane&15, hi = lane>>4) holds, from row pointer p = base + hi*8:
//   elements 0..7  = p[0..7]    (K = k0 + hi*8 .. +7)
//   elements 8..15 = p[16..23]  (K = k0 + 16 + hi*8 .. +7)
__device__ inline v16bf load_frag(const bf16* p) {
  Frag16 f;
  f.h.lo = *reinterpret_cast<const v8bf*>(p);
  f.h.hi = *reinterpret_cast<const v8bf*>(p + 16);
  return f.v;
}

__device__ inline v8f wmma_bf16(v16bf a, v16bf b, v8f c) {
  return __builtin_amdgcn_wmma_f32_16x16x32_bf16(
      /*neg_a=*/false, a, /*neg_b=*/false, b,
      /*c_mod=*/(short)0, c, /*reuse_a=*/false, /*reuse_b=*/false);
}

// ---------------------------------------------------------------------------
// Tensor Data Mover: 2D tile load (bf16 elements) global -> LDS.
// tile_d0 = contiguous elements per row, tile_d1 = rows, row stride in elems.
// D# per CDNA5 ISA (group0 + group1, groups 2/3 zero).
// This toolchain's builtin is the 6-arg form (g0, g1, g2, g3, extra, cpol).
// ---------------------------------------------------------------------------
__device__ inline void tdm_load_2d_bf16(u32 lds_off, const void* gaddr,
                                        u32 tile_d0, u32 tile_d1, u32 stride_elems)
{
  const unsigned long long ga = (unsigned long long)gaddr;
  v4u g0;
  g0[0] = 1u;                                            // count=1, user desc
  g0[1] = lds_off;                                       // lds_addr (bytes)
  g0[2] = (u32)ga;                                       // global_addr[31:0]
  g0[3] = ((u32)(ga >> 32) & 0x01FFFFFFu) | (2u << 30);  // ga[56:32] | type=2
  v8i g1;
  g1[0] = (int)(1u << 16);                               // data_size=1 -> 2B
  g1[1] = (int)((tile_d0 & 0xFFFFu) << 16);              // tensor_dim0[15:0]
  g1[2] = (int)(((tile_d0 >> 16) & 0xFFFFu) |            // tensor_dim0[31:16]
                ((tile_d1 & 0xFFFFu) << 16));            // tensor_dim1[15:0]
  g1[3] = (int)(((tile_d1 >> 16) & 0xFFFFu) |            // tensor_dim1[31:16]
                ((tile_d0 & 0xFFFFu) << 16));            // tile_dim0
  g1[4] = (int)(tile_d1 & 0xFFFFu);                      // tile_dim1 (tile_dim2=0)
  g1[5] = (int)stride_elems;                             // tensor_dim0_stride[31:0]
  g1[6] = 0;                                             // stride0[47:32]|stride1[15:0]
  g1[7] = 0;                                             // tensor_dim1_stride[47:16]
  const v4i z4 = {0, 0, 0, 0};
  const v8i z8 = {0, 0, 0, 0, 0, 0, 0, 0};
  __builtin_amdgcn_tensor_load_to_lds(g0, g1, z4, z4, z8, 0);
}

// ---------------------------------------------------------------------------
// LayerNorm: f32 rows -> bf16 rows. One block per row.
// ---------------------------------------------------------------------------
__global__ __launch_bounds__(256)
void ln_to_bf16_kernel(const float* __restrict__ x, const float* __restrict__ g,
                       const float* __restrict__ bb, bf16* __restrict__ y, int dim)
{
  __shared__ float redS[8], redQ[8];
  const int row = blockIdx.x;
  const float* xr = x + (size_t)row * dim;
  float s = 0.f, q = 0.f;
  for (int i = threadIdx.x; i < dim; i += 256) { float v = xr[i]; s += v; q += v * v; }
#pragma unroll
  for (int o = 16; o > 0; o >>= 1) { s += __shfl_xor(s, o, 32); q += __shfl_xor(q, o, 32); }
  const int w = threadIdx.x >> 5, l = threadIdx.x & 31;
  if (l == 0) { redS[w] = s; redQ[w] = q; }
  __syncthreads();
  s = redS[l & 7]; q = redQ[l & 7];
#pragma unroll
  for (int o = 4; o > 0; o >>= 1) { s += __shfl_xor(s, o, 32); q += __shfl_xor(q, o, 32); }
  const float inv_d = 1.f / (float)dim;
  const float mean = s * inv_d;
  const float rstd = rsqrtf(q * inv_d - mean * mean + 1e-5f);
  bf16* yr = y + (size_t)row * dim;
  for (int i = threadIdx.x; i < dim; i += 256)
    yr[i] = (bf16)((xr[i] - mean) * rstd * g[i] + bb[i]);
}

// ---------------------------------------------------------------------------
// Weight prep: W (K x N, f32, row-major) -> W^T (N x K, bf16).
// ---------------------------------------------------------------------------
__global__ __launch_bounds__(256)
void transpose_to_bf16(const float* __restrict__ W, bf16* __restrict__ WT, int K, int N)
{
  int idx = blockIdx.x * blockDim.x + threadIdx.x;
  if (idx >= K * N) return;
  int k = idx / N, n = idx - k * N;
  WT[(size_t)n * K + k] = (bf16)W[idx];
}

// ---------------------------------------------------------------------------
// WMMA GEMM: C[M x N] = A[M x K](bf16,rm) * B with B given as BT[N x K](bf16,rm).
// Wave tile 16x64, block 128x64, K-step 32, f32 accumulate, optional bias.
// All 5 fragment loads issued before the 4 WMMAs (one clause, one wait).
// MODE 0: bf16 row-major out. MODE 1: f32 row-major out.
// MODE 2: bf16 "V^T" out: Cout[(b*SD_ + n)*Tdim + t], m = b*Tdim + t.
// ---------------------------------------------------------------------------
template <int MODE>
__global__ __launch_bounds__(256)
void gemm_wmma(const bf16* __restrict__ A, int lda,
               const bf16* __restrict__ BT, int ldb,
               const float* __restrict__ bias,
               void* __restrict__ Cout, int ldc, int K, int Tdim)
{
  const int lane = threadIdx.x & 31;
  const int wave = threadIdx.x >> 5;
  const int low = lane & 15, hi = lane >> 4;
  const int n0 = blockIdx.x * 64;
  const int m0 = blockIdx.y * 128 + wave * 16;

  const bf16* ap = A  + (size_t)(m0 + low) * lda + hi * 8;
  const bf16* bp = BT + (size_t)(n0 + low) * ldb + hi * 8;
  const size_t bstep = (size_t)16 * ldb;

  v8f acc[4] = {};
  for (int k0 = 0; k0 < K; k0 += 32) {
    if (k0 + 128 < K) {
      __builtin_prefetch(ap + k0 + 128, 0, 0);              // global_prefetch_b8
      __builtin_prefetch(bp + k0 + 128, 0, 0);
    }
    const v16bf a  = load_frag(ap + k0);
    const v16bf b0 = load_frag(bp + k0);
    const v16bf b1 = load_frag(bp + bstep + k0);
    const v16bf b2 = load_frag(bp + 2 * bstep + k0);
    const v16bf b3 = load_frag(bp + 3 * bstep + k0);
    acc[0] = wmma_bf16(a, b0, acc[0]);
    acc[1] = wmma_bf16(a, b1, acc[1]);
    acc[2] = wmma_bf16(a, b2, acc[2]);
    acc[3] = wmma_bf16(a, b3, acc[3]);
  }

#pragma unroll
  for (int j = 0; j < 4; ++j) {
    const int ng = n0 + j * 16 + low;
    const float badd = bias ? bias[ng] : 0.f;
#pragma unroll
    for (int r = 0; r < 8; ++r) {
      const int mg = m0 + r + 8 * hi;   // C-frag: M = r + 8*hi, N = low (+16j)
      const float vv = acc[j][r] + badd;
      if (MODE == 0) {
        ((bf16*)Cout)[(size_t)mg * ldc + ng] = (bf16)vv;
      } else if (MODE == 1) {
        ((float*)Cout)[(size_t)mg * ldc + ng] = vv;
      } else {
        const int bidx = mg / Tdim, t = mg - bidx * Tdim;
        ((bf16*)Cout)[((size_t)bidx * SD_ + ng) * Tdim + t] = (bf16)vv;
      }
    }
  }
}

// ---------------------------------------------------------------------------
// Flash-style attention per (b,h). Block = 8 waves x 16 query rows = 128 rows.
// K/V chunks (32 t-steps) are TDM-DMA'd into LDS double buffers shared by all
// waves; QK^T and PV both WMMA; online softmax with lane-xor row reductions;
// P bounced through LDS (C-frag -> A-frag layout).
//   LDS map (assumed base 0): K bufs @0,@4096 (32t x 64d), V bufs @8192,@12288
//   (64d x 32t), P tiles @16384 (per-wave 16x40).
// ---------------------------------------------------------------------------
#define LDS_K0 0u
#define LDS_V0 8192u
#define LDS_P0 16384u

__global__ __launch_bounds__(256)
void attention_kernel(const bf16* __restrict__ Q, const bf16* __restrict__ Km,
                      const bf16* __restrict__ Vt, const int* __restrict__ tmask,
                      bf16* __restrict__ Ao)
{
  __shared__ __align__(16) char smem[16384 + 8 * 16 * 40 * 2];
  bf16* const Kbuf0 = (bf16*)(smem + LDS_K0);
  bf16* const Vbuf0 = (bf16*)(smem + LDS_V0);

  const int lane = threadIdx.x & 31;
  const int wave = threadIdx.x >> 5;
  const int low = lane & 15, hi = lane >> 4;
  const int b = blockIdx.x >> 3, h = blockIdx.x & 7;
  const int s0 = blockIdx.y * 128 + wave * 16;

  bf16* const Pl = (bf16*)(smem + LDS_P0) + wave * 16 * 40;

  const bf16* qp = Q + (size_t)(b * S_ + s0 + low) * SD_ + h * D_ + hi * 8;
  const v16bf qa0 = load_frag(qp);        // d = 0..31
  const v16bf qa1 = load_frag(qp + 32);   // d = 32..63

  v8f o0 = {}, o1 = {}, o2 = {}, o3 = {};
  float mr[8], lr[8];
#pragma unroll
  for (int r = 0; r < 8; ++r) { mr[r] = -3.4e38f; lr[r] = 0.f; }

  const bf16* kgbase = Km + (size_t)(b * T_) * SD_ + h * D_;        // [t][d] rows
  const bf16* vgbase = Vt + ((size_t)b * SD_ + h * D_) * T_;        // [d][t] rows
  const int*  mbase  = tmask + b * T_;

  // TDM prologue: chunk 0 into buffer 0 (issued by wave 0 only).
  if (threadIdx.x < 32) {
    tdm_load_2d_bf16(LDS_K0, kgbase, 64, 32, SD_);
    tdm_load_2d_bf16(LDS_V0, vgbase, 32, 64, T_);
  }

  const int NCH = T_ / 32;
  for (int i = 0; i < NCH; ++i) {
    const int t0 = i * 32;
    const int cur = i & 1, nxt = cur ^ 1;

    if (threadIdx.x < 32) {
      if (i + 1 < NCH) {
        tdm_load_2d_bf16(LDS_K0 + nxt * 4096u, kgbase + (size_t)(t0 + 32) * SD_, 64, 32, SD_);
        tdm_load_2d_bf16(LDS_V0 + nxt * 4096u, vgbase + (t0 + 32), 32, 64, T_);
        __builtin_amdgcn_s_wait_tensorcnt(2);   // chunk i landed; i+1 in flight
      } else {
        __builtin_amdgcn_s_wait_tensorcnt(0);
      }
    }
    __syncthreads();                            // publish buffer `cur` to all waves

    const bf16* Kl = Kbuf0 + cur * 2048;        // [32][64] bf16
    const bf16* Vl = Vbuf0 + cur * 2048;        // [64][32] bf16

    const bool okL = mbase[t0 + low] != 0;        // C-frag column N = low
    const bool okR = mbase[t0 + 16 + low] != 0;

    // All K fragments loaded up front -> single dscnt wait, 4 back-to-back WMMAs.
    const bf16* klp = Kl + (size_t)low * 64 + hi * 8;
    const v16bf kf0 = load_frag(klp);
    const v16bf kf1 = load_frag(klp + 32);
    const v16bf kf2 = load_frag(klp + 16 * 64);
    const v16bf kf3 = load_frag(klp + 16 * 64 + 32);
    v8f sl = {}, sr = {};
    sl = wmma_bf16(qa0, kf0, sl);
    sl = wmma_bf16(qa1, kf1, sl);
    sr = wmma_bf16(qa0, kf2, sr);
    sr = wmma_bf16(qa1, kf3, sr);

#pragma unroll
    for (int r = 0; r < 8; ++r) {
      float a = fminf(fmaxf(sl[r] * 0.125f, -50.f), 50.f);   // scale = D^-0.5
      float c = fminf(fmaxf(sr[r] * 0.125f, -50.f), 50.f);
      a = okL ? a : -3.4e38f;
      c = okR ? c : -3.4e38f;
      float red = fmaxf(a, c);
      red = fmaxf(red, __shfl_xor(red, 1, 32));
      red = fmaxf(red, __shfl_xor(red, 2, 32));
      red = fmaxf(red, __shfl_xor(red, 4, 32));
      red = fmaxf(red, __shfl_xor(red, 8, 32));
      const float nm  = fmaxf(mr[r], red);
      const float sce = __expf(mr[r] - nm);
      float el = okL ? __expf(a - nm) : 0.f;     // forced 0 when masked
      float er = okR ? __expf(c - nm) : 0.f;
      float rs = el + er;
      rs += __shfl_xor(rs, 1, 32);
      rs += __shfl_xor(rs, 2, 32);
      rs += __shfl_xor(rs, 4, 32);
      rs += __shfl_xor(rs, 8, 32);
      lr[r] = lr[r] * sce + rs;
      mr[r] = nm;
      o0[r] *= sce; o1[r] *= sce; o2[r] *= sce; o3[r] *= sce;
      Pl[(r + 8 * hi) * 40 + low]      = (bf16)el;           // C-frag -> LDS
      Pl[(r + 8 * hi) * 40 + low + 16] = (bf16)er;
    }

    asm volatile("s_wait_dscnt 0" ::: "memory");             // P stores -> reads

    const v16bf pa = load_frag(Pl + low * 40 + hi * 8);      // A-frag 16x32

    const bf16* vlp = Vl + (size_t)low * 32 + hi * 8;
    const v16bf vf0 = load_frag(vlp);
    const v16bf vf1 = load_frag(vlp + 16 * 32);
    const v16bf vf2 = load_frag(vlp + 32 * 32);
    const v16bf vf3 = load_frag(vlp + 48 * 32);
    o0 = wmma_bf16(pa, vf0, o0);
    o1 = wmma_bf16(pa, vf1, o1);
    o2 = wmma_bf16(pa, vf2, o2);
    o3 = wmma_bf16(pa, vf3, o3);

    __syncthreads();                            // done with buffer `cur`
  }

  bf16* op = Ao + (size_t)(b * S_ + s0 + 8 * hi) * SD_ + h * D_;
#pragma unroll
  for (int r = 0; r < 8; ++r) {
    const float inv = 1.f / fmaxf(lr[r], 1e-20f);            // fully-masked row -> 0
    bf16* orow = op + (size_t)r * SD_;
    orow[low]      = (bf16)(o0[r] * inv);
    orow[16 + low] = (bf16)(o1[r] * inv);
    orow[32 + low] = (bf16)(o2[r] * inv);
    orow[48 + low] = (bf16)(o3[r] * inv);
  }
}

// ---------------------------------------------------------------------------
// Final: out = source + alpha * LayerNorm(proj + bo); alpha = 0.25*(tanh+1).
// ---------------------------------------------------------------------------
__global__ __launch_bounds__(256)
void final_kernel(const float* __restrict__ proj, const float* __restrict__ bo,
                  const float* __restrict__ g, const float* __restrict__ bb,
                  const float* __restrict__ src, const float* __restrict__ alpha_raw,
                  float* __restrict__ out)
{
  __shared__ float redS[8], redQ[8];
  const int row = blockIdx.x;
  const int i0 = threadIdx.x, i1 = threadIdx.x + 256;
  const float* xr = proj + (size_t)row * SD_;
  const float v0 = xr[i0] + bo[i0];
  const float v1 = xr[i1] + bo[i1];
  float s = v0 + v1, q = v0 * v0 + v1 * v1;
#pragma unroll
  for (int o = 16; o > 0; o >>= 1) { s += __shfl_xor(s, o, 32); q += __shfl_xor(q, o, 32); }
  const int w = threadIdx.x >> 5, l = threadIdx.x & 31;
  if (l == 0) { redS[w] = s; redQ[w] = q; }
  __syncthreads();
  s = redS[l & 7]; q = redQ[l & 7];
#pragma unroll
  for (int o = 4; o > 0; o >>= 1) { s += __shfl_xor(s, o, 32); q += __shfl_xor(q, o, 32); }
  const float mean = s * (1.f / SD_);
  const float rstd = rsqrtf(q * (1.f / SD_) - mean * mean + 1e-5f);
  const float alpha = 0.25f * (tanhf(alpha_raw[0]) + 1.f);
  const float* srow = src + (size_t)row * SD_;
  float* orow = out + (size_t)row * SD_;
  orow[i0] = srow[i0] + alpha * ((v0 - mean) * rstd * g[i0] + bb[i0]);
  orow[i1] = srow[i1] + alpha * ((v1 - mean) * rstd * g[i1] + bb[i1]);
}

// ---------------------------------------------------------------------------
extern "C" void kernel_launch(void* const* d_in, const int* in_sizes, int n_in,
                              void* d_out, int out_size, void* d_ws, size_t ws_size,
                              hipStream_t stream)
{
  const float* source = (const float*)d_in[0];
  const float* target = (const float*)d_in[1];
  const int*   tmask  = (const int*)d_in[2];
  const float* ns_g = (const float*)d_in[3];
  const float* ns_b = (const float*)d_in[4];
  const float* nt_g = (const float*)d_in[5];
  const float* nt_b = (const float*)d_in[6];
  const float* no_g = (const float*)d_in[7];
  const float* no_b = (const float*)d_in[8];
  const float* Wq = (const float*)d_in[9];   const float* bq = (const float*)d_in[10];
  const float* Wk = (const float*)d_in[11];  const float* bk = (const float*)d_in[12];
  const float* Wv = (const float*)d_in[13];  const float* bv = (const float*)d_in[14];
  const float* Wo = (const float*)d_in[15];  const float* bo = (const float*)d_in[16];
  const float* alpha_raw = (const float*)d_in[17];
  (void)in_sizes; (void)n_in; (void)out_size; (void)ws_size;

  char* ws = (char*)d_ws;
  size_t off = 0;
  auto take = [&](size_t bytes) -> char* {
    char* p = ws + off;
    off += (bytes + 255) & ~(size_t)255;
    return p;
  };
  bf16*  srcN = (bf16*)take((size_t)B_ * S_ * SD_ * 2);
  bf16*  tgtN = (bf16*)take((size_t)B_ * T_ * TD_ * 2);
  bf16*  WqT  = (bf16*)take((size_t)SD_ * SD_ * 2);
  bf16*  WkT  = (bf16*)take((size_t)TD_ * SD_ * 2);
  bf16*  WvT  = (bf16*)take((size_t)TD_ * SD_ * 2);
  bf16*  WoT  = (bf16*)take((size_t)SD_ * SD_ * 2);
  bf16*  Qb   = (bf16*)take((size_t)B_ * S_ * SD_ * 2);
  bf16*  Kb   = (bf16*)take((size_t)B_ * T_ * SD_ * 2);
  bf16*  Vtb  = (bf16*)take((size_t)B_ * T_ * SD_ * 2);
  bf16*  AoB  = (bf16*)take((size_t)B_ * S_ * SD_ * 2);
  float* proj = (float*)take((size_t)B_ * S_ * SD_ * 4);

  // 1) LayerNorms (f32 -> bf16)
  ln_to_bf16_kernel<<<B_ * S_, 256, 0, stream>>>(source, ns_g, ns_b, srcN, SD_);
  ln_to_bf16_kernel<<<B_ * T_, 256, 0, stream>>>(target, nt_g, nt_b, tgtN, TD_);

  // 2) Weight prep (f32 KxN -> bf16 NxK)
  transpose_to_bf16<<<(SD_ * SD_ + 255) / 256, 256, 0, stream>>>(Wq, WqT, SD_, SD_);
  transpose_to_bf16<<<(TD_ * SD_ + 255) / 256, 256, 0, stream>>>(Wk, WkT, TD_, SD_);
  transpose_to_bf16<<<(TD_ * SD_ + 255) / 256, 256, 0, stream>>>(Wv, WvT, TD_, SD_);
  transpose_to_bf16<<<(SD_ * SD_ + 255) / 256, 256, 0, stream>>>(Wo, WoT, SD_, SD_);

  // 3) Projections (WMMA): Q row-major, K row-major, V stored transposed (DxT)
  gemm_wmma<0><<<dim3(SD_ / 64, (B_ * S_) / 128), 256, 0, stream>>>(
      srcN, SD_, WqT, SD_, bq, Qb, SD_, SD_, 0);
  gemm_wmma<0><<<dim3(SD_ / 64, (B_ * T_) / 128), 256, 0, stream>>>(
      tgtN, TD_, WkT, TD_, bk, Kb, SD_, TD_, 0);
  gemm_wmma<2><<<dim3(SD_ / 64, (B_ * T_) / 128), 256, 0, stream>>>(
      tgtN, TD_, WvT, TD_, bv, Vtb, 0, TD_, T_);

  // 4) Attention (TDM-staged K/V + WMMA QK^T / PV + online softmax)
  attention_kernel<<<dim3(B_ * H_, S_ / 128), 256, 0, stream>>>(Qb, Kb, Vtb, tmask, AoB);

  // 5) Output projection (WMMA, f32 out)
  gemm_wmma<1><<<dim3(SD_ / 64, (B_ * S_) / 128), 256, 0, stream>>>(
      AoB, SD_, WoT, SD_, nullptr, proj, SD_, SD_, 0);

  // 6) Final LN + gated residual
  final_kernel<<<B_ * S_, 256, 0, stream>>>(proj, bo, no_g, no_b, source, alpha_raw,
                                            (float*)d_out);
}
